// TSPModel_28707561406869
// MI455X (gfx1250) — matrix operand order repaired and checked by reading, not compiled, not run on hardware
//
#include <hip/hip_runtime.h>
#include <cstdint>
#include <cstddef>

// ---------------- problem constants ----------------
#define T_TOK   131072              // B*N tokens
#define DDIM    256
#define HDIM    512
#define NEXP    8
#define TOPK    2
#define RBLK    32                  // router tokens per block
#define NRBLK   (T_TOK / RBLK)      // 4096
#define GM      128                 // gemm rows per block
#define CAP     (2*T_TOK + NEXP*GM) // padded slot capacity = 263168
#define XS2     260                 // LDS row stride (bf16) for x tile
#define HS2     132                 // LDS row stride (bf16) for h chunk

typedef __attribute__((ext_vector_type(16))) __bf16 v16bf;
typedef __attribute__((ext_vector_type(8)))  float  v8f;
typedef __attribute__((ext_vector_type(4))) unsigned int u32x4;
typedef __attribute__((ext_vector_type(8))) int i32x8;
typedef __attribute__((ext_vector_type(4))) int i32x4;

union Frag16 { uint32_t u[8]; v16bf v; };

__device__ __forceinline__ uint16_t f32_to_bf16(float f) {
    uint32_t u = __float_as_uint(f);
    uint32_t r = u + 0x7FFFu + ((u >> 16) & 1u);   // round-to-nearest-even
    return (uint16_t)(r >> 16);
}
__device__ __forceinline__ float bf16_to_f32(uint16_t v) {
    return __uint_as_float((uint32_t)v << 16);
}

// ---------------- TDM: async 2D tile load (global -> LDS, with row pad) -----
// Loads a GM x 256 bf16 tile whose rows are contiguous in memory into LDS at
// stride XS2 (=256+4) via the LDS-padding feature:
//   pad_interval=6 -> pad after every 128 DWORDs (256 bf16 = one row)
//   pad_amount=1   -> insert 2 DWORDs (4 bf16) of padding
#if defined(__has_include)
#  if __has_include(<hip/amd_detail/amd_gfx1250_TDM.h>)
#    define TDM_ARGS6 1
#  endif
#endif

__device__ __forceinline__ void tdm_load_x_tile(const uint16_t* gsrc,
                                                uint32_t lds_off) {
    uint64_t ga = (uint64_t)(uintptr_t)gsrc;
    // D# group0: count=1, lds_addr, 57-bit global addr, type=2 ("image")
    u32x4 g0 = { 1u, lds_off, (uint32_t)ga,
                 (uint32_t)((ga >> 32) & 0x01FFFFFFu) | (2u << 30) };
    // D# group1: data_size=2B, pad_enable, pad_interval=6, pad_amount=1,
    // tensor_dim0=256, tensor_dim1=GM, tile_dim0=256, tile_dim1=GM,
    // tensor_dim0_stride=256
    i32x8 g1 = { (int)((1u << 16) | (1u << 20) | (6u << 22) | (1u << 25)),
                 (int)(256u << 16),            // tensor_dim0[15:0] @ bits 63:48
                 (int)((unsigned)GM << 16),    // tensor_dim1[15:0] @ bits 95:80
                 (int)(256u << 16),            // tile_dim0 @ bits 127:112
                 GM,                           // tile_dim1 @ bits 143:128
                 256,                          // tensor_dim0_stride[31:0]
                 0, 0 };
    i32x4 gz = { 0, 0, 0, 0 };                 // groups 2/3 unused (2D tensor)
#ifdef TDM_ARGS6
    i32x8 gz8 = { 0, 0, 0, 0, 0, 0, 0, 0 };
    __builtin_amdgcn_tensor_load_to_lds(g0, g1, gz, gz, gz8, 0);
#else
    __builtin_amdgcn_tensor_load_to_lds(g0, g1, gz, gz, 0);
#endif
}

// ---------------- kernel: zero control block ----------------
__global__ void zero_k(unsigned* __restrict__ ctrl) {
    if (threadIdx.x < 8) ctrl[threadIdx.x] = 0u;   // counts
}

// ---------------- kernel: weight transpose + bf16 convert ----------------
// w1T[e][h][d], w2T[e][d][h], sw1T[h][d], sw2T[d][h]  (B operand wants [N][K])
__global__ __launch_bounds__(256) void prep_weights_k(
    const float* __restrict__ w1, const float* __restrict__ w2,
    const float* __restrict__ sw1, const float* __restrict__ sw2,
    uint16_t* __restrict__ w1T, uint16_t* __restrict__ w2T,
    uint16_t* __restrict__ sw1T, uint16_t* __restrict__ sw2T)
{
    const int HD = HDIM * DDIM;           // 131072
    const int total = 18 * HD;
    for (int i = blockIdx.x * blockDim.x + threadIdx.x; i < total;
         i += gridDim.x * blockDim.x) {
        if (i < 8 * HD) {
            int e = i / HD, r = i % HD, h = r / DDIM, d = r % DDIM;
            w1T[i] = f32_to_bf16(w1[((size_t)e * DDIM + d) * HDIM + h]);
        } else if (i < 16 * HD) {
            int j = i - 8 * HD;
            int e = j / HD, r = j % HD, d = r / HDIM, h = r % HDIM;
            w2T[j] = f32_to_bf16(w2[((size_t)e * HDIM + h) * DDIM + d]);
        } else if (i < 17 * HD) {
            int j = i - 16 * HD; int h = j / DDIM, d = j % DDIM;
            sw1T[j] = f32_to_bf16(sw1[(size_t)d * HDIM + h]);
        } else {
            int j = i - 17 * HD; int d = j / HDIM, h = j % HDIM;
            sw2T[j] = f32_to_bf16(sw2[(size_t)h * DDIM + d]);
        }
    }
}

// ---------------- kernel: router (softmax/top-2) + x->bf16 ----------------
__global__ __launch_bounds__(256) void router_k(
    const float* __restrict__ x, const float* __restrict__ rw,
    const float* __restrict__ rb,
    uint16_t* __restrict__ xb, int* __restrict__ sel01,
    float* __restrict__ wpair, unsigned* __restrict__ counts,
    float* __restrict__ partial)
{
    __shared__ float xt[RBLK * XS2];
    __shared__ float rwl[DDIM * NEXP];
    __shared__ float sprob[RBLK * NEXP];
    __shared__ int   ssel[RBLK * 2];

    const int tid = threadIdx.x;
    const size_t base = (size_t)blockIdx.x * RBLK * DDIM;

    for (int i = tid; i < DDIM * NEXP; i += 256) rwl[i] = rw[i];
    for (int i = tid; i < RBLK * DDIM; i += 256) {
        int row = i >> 8, col = i & 255;
        float v = x[base + i];
        xt[row * XS2 + col] = v;
        xb[base + i] = f32_to_bf16(v);
    }
    __syncthreads();

    if (tid < RBLK) {
        float acc[NEXP];
        #pragma unroll
        for (int e = 0; e < NEXP; ++e) acc[e] = rb[e];
        for (int d = 0; d < DDIM; ++d) {
            float xv = xt[tid * XS2 + d];
            #pragma unroll
            for (int e = 0; e < NEXP; ++e) acc[e] += xv * rwl[d * NEXP + e];
        }
        float mx = acc[0];
        #pragma unroll
        for (int e = 1; e < NEXP; ++e) mx = acc[e] > mx ? acc[e] : mx;
        float p[NEXP]; float s = 0.f;
        #pragma unroll
        for (int e = 0; e < NEXP; ++e) { p[e] = __expf(acc[e] - mx); s += p[e]; }
        float inv = 1.0f / s;
        #pragma unroll
        for (int e = 0; e < NEXP; ++e) p[e] *= inv;
        int i0 = 0; float p0 = p[0];
        #pragma unroll
        for (int e = 1; e < NEXP; ++e) if (p[e] > p0) { p0 = p[e]; i0 = e; }
        int i1 = -1; float p1 = -1.0f;
        #pragma unroll
        for (int e = 0; e < NEXP; ++e)
            if (e != i0 && p[e] > p1) { p1 = p[e]; i1 = e; }
        float wsum = p0 + p1 + 1e-9f;
        size_t t = (size_t)blockIdx.x * RBLK + tid;
        sel01[t] = i0 | (i1 << 8);
        wpair[2 * t]     = p0 / wsum;
        wpair[2 * t + 1] = p1 / wsum;
        #pragma unroll
        for (int e = 0; e < NEXP; ++e) sprob[tid * NEXP + e] = p[e];
        ssel[tid * 2] = i0; ssel[tid * 2 + 1] = i1;
    }
    __syncthreads();
    if (tid < NEXP) {   // deterministic per-block partials for aux loss
        float si = 0.f; int sl = 0;
        for (int m = 0; m < RBLK; ++m) {
            si += sprob[m * NEXP + tid];
            sl += (ssel[m * 2] == tid ? 1 : 0) + (ssel[m * 2 + 1] == tid ? 1 : 0);
        }
        partial[(size_t)blockIdx.x * 16 + tid]     = si;
        partial[(size_t)blockIdx.x * 16 + 8 + tid] = (float)sl;
        atomicAdd(&counts[tid], (unsigned)sl);     // integer: deterministic
    }
}

// ---------------- kernel: segment offsets (pad to GM) ----------------
__global__ void offsets_k(const unsigned* __restrict__ cnt,
                          unsigned* __restrict__ basep,
                          unsigned* __restrict__ cursor) {
    if (threadIdx.x == 0) {
        unsigned b = 0;
        for (int e = 0; e < NEXP; ++e) {
            basep[e] = b; cursor[e] = b;
            b += (cnt[e] + (GM - 1)) & ~(unsigned)(GM - 1);
        }
        basep[NEXP] = b;
    }
}

// ---------------- kernel: init slot arrays (pad slots -> token 0, scale 0) --
__global__ __launch_bounds__(256) void init_slots_k(int* __restrict__ tlist,
                                                    float* __restrict__ sscale) {
    int i = blockIdx.x * 256 + threadIdx.x;
    if (i < CAP) { tlist[i] = 0; sscale[i] = 0.0f; }
}

// ---------------- kernel: scatter tokens into expert segments ----------------
__global__ __launch_bounds__(256) void scatter_k(
    const int* __restrict__ sel01, const float* __restrict__ wpair,
    unsigned* __restrict__ cursor, int* __restrict__ tlist,
    float* __restrict__ sscale, int* __restrict__ slot0, int* __restrict__ slot1)
{
    int t = blockIdx.x * 256 + threadIdx.x;
    if (t >= T_TOK) return;
    int s = sel01[t];
    int e0 = s & 0xFF, e1 = (s >> 8) & 0xFF;
    unsigned p0 = atomicAdd(&cursor[e0], 1u);
    unsigned p1 = atomicAdd(&cursor[e1], 1u);
    tlist[p0] = t; sscale[p0] = wpair[2 * t];
    tlist[p1] = t; sscale[p1] = wpair[2 * t + 1];
    slot0[t] = (int)p0; slot1[t] = (int)p1;
}

// ---------------- kernel: gathered expert GEMM (bf16 WMMA) ----------------
// blockIdx.y = expert (0..7 routed -> ybuf, 8 = shared -> out directly)
__global__ __launch_bounds__(256) void moe_gemm_k(
    const uint16_t* __restrict__ xb,
    const uint16_t* __restrict__ w1T, const uint16_t* __restrict__ w2T,
    const uint16_t* __restrict__ sw1T, const uint16_t* __restrict__ sw2T,
    const float* __restrict__ b1, const float* __restrict__ b2,
    const float* __restrict__ sb1, const float* __restrict__ sb2,
    const int* __restrict__ tlist, const float* __restrict__ sscale,
    const unsigned* __restrict__ cnt, const unsigned* __restrict__ basep,
    uint16_t* __restrict__ ybuf, float* __restrict__ out)
{
    const int e = blockIdx.y;
    const bool se = (e == NEXP);
    const int bx = blockIdx.x;
    if (!se && (unsigned)bx * GM >= cnt[e]) return;   // uniform exit

    __shared__ uint16_t xt[GM * XS2];   // 128 x 256 bf16 (gathered) rows
    __shared__ uint16_t hct[GM * HS2];  // 128 x 128 bf16 hidden chunk
    __shared__ float    srow[GM];       // per-row combine weight (folded)
    __shared__ int      tl[GM];

    const int tid = threadIdx.x;
    const int lane = tid & 31, wave = tid >> 5;
    const int grp = lane >> 4, l16 = lane & 15;
    const unsigned rowbase = se ? (unsigned)bx * GM : basep[e] + (unsigned)bx * GM;

    if (tid < GM) {
        if (se) { srow[tid] = 1.0f; }
        else    { tl[tid] = tlist[rowbase + tid]; srow[tid] = sscale[rowbase + tid]; }
    }
    if (se) {
        // contiguous rows: stage via Tensor Data Mover (async DMA, LDS pad)
        if (wave == 0) {
            tdm_load_x_tile(xb + (size_t)bx * GM * DDIM, (uint32_t)(size_t)&xt[0]);
            __builtin_amdgcn_s_wait_tensorcnt(0);
        }
    } else {
        __syncthreads();   // tl ready
        for (int c = tid; c < GM * (DDIM / 4); c += 256) {   // 8B chunks
            int row = c >> 6, col4 = (c & 63) << 2;
            *(uint64_t*)&xt[row * XS2 + col4] =
                *(const uint64_t*)&xb[(size_t)tl[row] * DDIM + col4];
        }
    }
    __syncthreads();

    const uint16_t* W1 = se ? sw1T : (w1T + (size_t)e * HDIM * DDIM);
    const uint16_t* W2 = se ? sw2T : (w2T + (size_t)e * DDIM * HDIM);
    const float* B1 = se ? sb1 : (b1 + e * HDIM);
    const float* B2 = se ? sb2 : (b2 + e * DDIM);

    v8f gacc[4][4];
    #pragma unroll
    for (int j = 0; j < 4; ++j)
        #pragma unroll
        for (int i = 0; i < 4; ++i) gacc[j][i] = {};

    const int m1 = wave >> 2, n1 = wave & 3;   // GEMM1: 4 mtiles x 2 ntiles
    const int m2 = wave & 1,  n2 = wave >> 1;  // GEMM2: 4 mtiles x 4 ntiles

    for (int hc = 0; hc < HDIM / 128; ++hc) {
        __builtin_prefetch(&W2[((size_t)(n2 * 64) + l16) * HDIM + hc * 128], 0, 1);
        // ---- GEMM1: Hc[128,128] = scale * relu(X @ W1[:,hc*128..] + b1) ----
        v8f c1[4][2];
        #pragma unroll
        for (int j = 0; j < 4; ++j) { c1[j][0] = {}; c1[j][1] = {}; }
        for (int kk = 0; kk < DDIM / 32; ++kk) {
            const int kb = kk * 32;
            Frag16 bf[2];
            #pragma unroll
            for (int i = 0; i < 2; ++i) {
                const int hcol = hc * 128 + (n1 * 2 + i) * 16 + l16;
                #pragma unroll
                for (int r = 0; r < 8; ++r)
                    bf[i].u[r] = *(const uint32_t*)&W1[(size_t)hcol * DDIM + kb + ((r < 4) ? (grp * 8 + 2 * r) : (16 + grp * 8 + 2 * (r - 4)))];
            }
            #pragma unroll
            for (int j = 0; j < 4; ++j) {
                Frag16 af;
                const int arow = (m1 * 4 + j) * 16 + l16;
                #pragma unroll
                for (int ii = 0; ii < 8; ++ii) {
                    int k = kb + ((ii < 4) ? (grp * 8 + 2 * ii)
                                           : (16 + grp * 8 + 2 * (ii - 4)));
                    af.u[ii] = *(const uint32_t*)&xt[arow * XS2 + k];
                }
                #pragma unroll
                for (int i = 0; i < 2; ++i)
                    c1[j][i] = __builtin_amdgcn_wmma_f32_16x16x32_bf16(
                        false, af.v, false, bf[i].v, (short)0, c1[j][i], false, false);
            }
        }
        #pragma unroll
        for (int i = 0; i < 2; ++i) {
            const int hcl = (n1 * 2 + i) * 16 + l16;
            const float bv = B1[hc * 128 + hcl];
            #pragma unroll
            for (int j = 0; j < 4; ++j) {
                #pragma unroll
                for (int r = 0; r < 8; ++r) {
                    int m = (m1 * 4 + j) * 16 + grp * 8 + r;
                    float v = c1[j][i][r] + bv;
                    v = v > 0.0f ? v : 0.0f;
                    v *= srow[m];                      // fold combine weight
                    hct[m * HS2 + hcl] = f32_to_bf16(v);
                }
            }
        }
        __syncthreads();
        // ---- GEMM2: gacc[128,256] += Hc @ W2[hc*128.., :] ----
        for (int kk2 = 0; kk2 < 4; ++kk2) {
            const int kb = kk2 * 32;
            Frag16 bf2[4];
            #pragma unroll
            for (int i = 0; i < 4; ++i) {
                const int dcol = (n2 * 4 + i) * 16 + l16;
                #pragma unroll
                for (int r = 0; r < 8; ++r)
                    bf2[i].u[r] = *(const uint32_t*)&W2[(size_t)dcol * HDIM + hc * 128 + kb + ((r < 4) ? (grp * 8 + 2 * r) : (16 + grp * 8 + 2 * (r - 4)))];
            }
            #pragma unroll
            for (int j = 0; j < 4; ++j) {
                Frag16 af;
                const int arow = (m2 * 4 + j) * 16 + l16;
                #pragma unroll
                for (int ii = 0; ii < 8; ++ii) {
                    int k = kb + ((ii < 4) ? (grp * 8 + 2 * ii)
                                           : (16 + grp * 8 + 2 * (ii - 4)));
                    af.u[ii] = *(const uint32_t*)&hct[arow * HS2 + k];
                }
                #pragma unroll
                for (int i = 0; i < 4; ++i)
                    gacc[j][i] = __builtin_amdgcn_wmma_f32_16x16x32_bf16(
                        false, af.v, false, bf2[i].v, (short)0, gacc[j][i], false, false);
            }
        }
        __syncthreads();
    }

    // ---- epilogue: + scale*b2; routed -> ybuf (bf16), shared -> out (f32) ----
    #pragma unroll
    for (int i = 0; i < 4; ++i) {
        const int dcol = (n2 * 4 + i) * 16 + l16;
        const float bv2 = B2[dcol];
        #pragma unroll
        for (int j = 0; j < 4; ++j) {
            #pragma unroll
            for (int r = 0; r < 8; ++r) {
                int m = (m2 * 4 + j) * 16 + grp * 8 + r;
                float v = gacc[j][i][r] + srow[m] * bv2;
                if (se) out[(size_t)(rowbase + m) * DDIM + dcol] = v;
                else    ybuf[(size_t)(rowbase + m) * DDIM + dcol] = f32_to_bf16(v);
            }
        }
    }
}

// ---------------- kernel: per-token combine (deterministic order) ----------
__global__ __launch_bounds__(256) void combine_k(
    const uint16_t* __restrict__ ybuf, const int* __restrict__ slot0,
    const int* __restrict__ slot1, float* __restrict__ out)
{
    int g = blockIdx.x * 256 + threadIdx.x;        // over T*64 float4 groups
    int t = g >> 6, c4 = (g & 63) << 2;
    int s0 = slot0[t], s1 = slot1[t];
    const uint16_t* y0 = &ybuf[(size_t)s0 * DDIM + c4];
    const uint16_t* y1 = &ybuf[(size_t)s1 * DDIM + c4];
    float* op = &out[(size_t)t * DDIM + c4];
    float4 o = *(float4*)op;
    o.x += bf16_to_f32(y0[0]) + bf16_to_f32(y1[0]);
    o.y += bf16_to_f32(y0[1]) + bf16_to_f32(y1[1]);
    o.z += bf16_to_f32(y0[2]) + bf16_to_f32(y1[2]);
    o.w += bf16_to_f32(y0[3]) + bf16_to_f32(y1[3]);
    *(float4*)op = o;
}

// ---------------- kernel: aux loss finalize (deterministic) ----------------
__global__ __launch_bounds__(64) void aux_k(const float* __restrict__ partial,
                                            float* __restrict__ out_aux)
{
    __shared__ float acc[16];
    int tid = threadIdx.x;
    if (tid < 16) {
        float s = 0.f;
        for (int b = 0; b < NRBLK; ++b) s += partial[(size_t)b * 16 + tid];
        acc[tid] = s;
    }
    __syncthreads();
    if (tid == 0) {
        float s = 0.f;
        #pragma unroll
        for (int e = 0; e < NEXP; ++e)
            s += (acc[e] / (float)T_TOK) * (acc[8 + e] / (float)(T_TOK * TOPK));
        out_aux[0] = (float)NEXP * s;
    }
}

// ---------------- host launcher ----------------
extern "C" void kernel_launch(void* const* d_in, const int* in_sizes, int n_in,
                              void* d_out, int out_size, void* d_ws, size_t ws_size,
                              hipStream_t stream) {
    (void)in_sizes; (void)n_in; (void)out_size; (void)ws_size;
    const float* x        = (const float*)d_in[0];
    const float* router_w = (const float*)d_in[1];
    const float* router_b = (const float*)d_in[2];
    const float* w1       = (const float*)d_in[3];
    const float* b1       = (const float*)d_in[4];
    const float* w2       = (const float*)d_in[5];
    const float* b2       = (const float*)d_in[6];
    const float* sw1      = (const float*)d_in[7];
    const float* sb1      = (const float*)d_in[8];
    const float* sw2      = (const float*)d_in[9];
    const float* sb2      = (const float*)d_in[10];
    float* out = (float*)d_out;

    char* ws = (char*)d_ws;
    size_t off = 0;
    uint16_t* xb    = (uint16_t*)(ws + off); off += (size_t)T_TOK * DDIM * 2;
    uint16_t* w1T   = (uint16_t*)(ws + off); off += (size_t)NEXP * HDIM * DDIM * 2;
    uint16_t* w2T   = (uint16_t*)(ws + off); off += (size_t)NEXP * DDIM * HDIM * 2;
    uint16_t* sw1T  = (uint16_t*)(ws + off); off += (size_t)HDIM * DDIM * 2;
    uint16_t* sw2T  = (uint16_t*)(ws + off); off += (size_t)DDIM * HDIM * 2;
    uint16_t* ybuf  = (uint16_t*)(ws + off); off += (size_t)CAP * DDIM * 2;
    int*      tlist = (int*)     (ws + off); off += (size_t)CAP * 4;
    float*    sscal = (float*)   (ws + off); off += (size_t)CAP * 4;
    int*      slot0 = (int*)     (ws + off); off += (size_t)T_TOK * 4;
    int*      slot1 = (int*)     (ws + off); off += (size_t)T_TOK * 4;
    int*      sel01 = (int*)     (ws + off); off += (size_t)T_TOK * 4;
    float*    wpair = (float*)   (ws + off); off += (size_t)T_TOK * 2 * 4;
    float*    part  = (float*)   (ws + off); off += (size_t)NRBLK * 16 * 4;
    unsigned* ctrl  = (unsigned*)(ws + off); off += 256;
    unsigned* cnt    = ctrl;          // [8]
    unsigned* basep  = ctrl + 16;     // [9]
    unsigned* cursor = ctrl + 32;     // [8]

    zero_k<<<1, 32, 0, stream>>>(cnt);
    prep_weights_k<<<1024, 256, 0, stream>>>(w1, w2, sw1, sw2, w1T, w2T, sw1T, sw2T);
    router_k<<<NRBLK, 256, 0, stream>>>(x, router_w, router_b, xb, sel01, wpair, cnt, part);
    offsets_k<<<1, 32, 0, stream>>>(cnt, basep, cursor);
    init_slots_k<<<CAP / 256, 256, 0, stream>>>(tlist, sscal);
    scatter_k<<<T_TOK / 256, 256, 0, stream>>>(sel01, wpair, cursor, tlist, sscal, slot0, slot1);
    moe_gemm_k<<<dim3(T_TOK / GM, NEXP + 1), 256, 0, stream>>>(
        xb, w1T, w2T, sw1T, sw2T, b1, b2, sb1, sb2,
        tlist, sscal, cnt, basep, ybuf, out);
    combine_k<<<(T_TOK * (DDIM / 4)) / 256, 256, 0, stream>>>(ybuf, slot0, slot1, out);
    aux_k<<<1, 64, 0, stream>>>(part, out + (size_t)T_TOK * DDIM);
}
